// Net_78907139162254
// MI455X (gfx1250) — matrix operand order repaired and checked by reading, not compiled, not run on hardware
//
#include <hip/hip_runtime.h>
#include <hip/hip_bf16.h>
#include <math.h>

#define B_SZ 4
#define SEQ_L 2048
#define IN_DIM 32
#define D_MODEL 256
#define N_LAYERS 2
#define ED 512
#define N_STATE 16
#define DT_RANK 16
#define D_CONV 4
#define NTOK (B_SZ * SEQ_L)   /* 8192 tokens */
#define MT 4                  /* M-tiles (16 rows each) per wave in the GEMM */

typedef __attribute__((ext_vector_type(16))) _Float16 v16h;
typedef __attribute__((ext_vector_type(8)))  _Float16 v8h;
typedef __attribute__((ext_vector_type(8)))  float    v8f;

union V16U { v16h v; v8h p[2]; };

__device__ __forceinline__ float sigmoid_f(float v) { return 1.f / (1.f + expf(-v)); }
__device__ __forceinline__ float softplus_f(float v) { return v > 20.f ? v : log1pf(expf(v)); }

// ---------------------------------------------------------------------------
// WMMA GEMM:  C(MxN) = A(MxK,f16) * W(NxK,f16)^T  (+ epilogue)
// One wave computes a 64x16 tile (4 M-subtiles sharing each B fragment):
// per k-step: 1 B-frag load feeds 4 v_wmma_f32_16x16x32_f16.
//   MODE 0: C = AB            MODE 1: C = AB + bias[col]
//   MODE 2: C = softplus(AB + bias[col])   MODE 3: C += AB (residual)
// Requires M % 64 == 0, N % 16 == 0, K % 32 == 0 (true for all call sites).
// ---------------------------------------------------------------------------
template <int MODE>
__global__ void wmma_gemm_kernel(const _Float16* __restrict__ A,
                                 const _Float16* __restrict__ W,
                                 float* __restrict__ C,
                                 const float* __restrict__ bias,
                                 int M, int N, int K) {
  const int ntN   = N >> 4;
  const int wave  = (blockIdx.x * blockDim.x + threadIdx.x) >> 5;
  const int lane  = threadIdx.x & 31;
  const int tileMB = wave / ntN;               // block of MT m-tiles
  const int tileN  = wave - tileMB * ntN;
  if (tileMB * (16 * MT) >= M) return;         // wave-uniform: EXEC stays all-1s

  const int r16 = lane & 15;                   // A row-in-tile == B col-in-tile
  const int hs  = lane >> 4;
  // A frag: lane holds row r16, K-halves {hs*8 .. +7} and {hs*8+16 .. +23}
  const _Float16* Arow[MT];
  #pragma unroll
  for (int i = 0; i < MT; ++i)
    Arow[i] = A + (size_t)((tileMB * MT + i) * 16 + r16) * K + hs * 8;
  // B frag (= W^T): lane holds W row r16, K-halves {hs*8 .. +7} and {hs*8+16 .. +23}
  const _Float16* Wrow = W + (size_t)(tileN * 16 + r16) * K + hs * 8;

  v8f acc[MT] = {};
  for (int k0 = 0; k0 < K; k0 += 32) {
    V16U b;
    b.p[0] = *(const v8h*)(Wrow + k0);
    b.p[1] = *(const v8h*)(Wrow + k0 + 16);
    #pragma unroll
    for (int i = 0; i < MT; ++i) {
      V16U a;
      a.p[0] = *(const v8h*)(Arow[i] + k0);
      a.p[1] = *(const v8h*)(Arow[i] + k0 + 16);
      acc[i] = __builtin_amdgcn_wmma_f32_16x16x32_f16(false, a.v, false, b.v,
                                                      (short)0, acc[i], false, false);
    }
  }

  // D layout: lane = col n (l&15); VGPR r = row hs*8 + r
  const int   ncol = tileN * 16 + r16;
  const float bn   = (MODE == 1 || MODE == 2) ? bias[ncol] : 0.f;
  #pragma unroll
  for (int i = 0; i < MT; ++i) {
    const int mbase = (tileMB * MT + i) * 16 + hs * 8;
    #pragma unroll
    for (int r = 0; r < 8; ++r) {
      float v = acc[i][r];
      float* dst = C + (size_t)(mbase + r) * N + ncol;
      if (MODE == 0) *dst = v;
      if (MODE == 1) *dst = v + bn;
      if (MODE == 2) *dst = softplus_f(v + bn);
      if (MODE == 3) *dst += v;
    }
  }
}

// --------------------------- elementwise helpers ---------------------------
__global__ void f32_to_f16_kernel(const float* __restrict__ s, _Float16* __restrict__ d, int n) {
  int i = blockIdx.x * blockDim.x + threadIdx.x;
  if (i < n) d[i] = (_Float16)s[i];
}

// Pad (rows x kin) f32 -> (rows x kout) f16 with zero columns kin..kout-1.
__global__ void pad_f16_kernel(const float* __restrict__ s, _Float16* __restrict__ d,
                               int rows, int kin, int kout) {
  int g = blockIdx.x * blockDim.x + threadIdx.x;
  if (g >= rows * kout) return;
  int r = g % kout, row = g / kout;
  d[g] = (r < kin) ? (_Float16)s[row * kin + r] : (_Float16)0.f;
}

// dt slice of dbc (48-wide) -> f16 padded to 32 cols
__global__ void dtprep_kernel(const float* __restrict__ dbc, _Float16* __restrict__ dt16) {
  int g = blockIdx.x * blockDim.x + threadIdx.x;
  if (g >= NTOK * 32) return;
  int r = g & 31, tok = g >> 5;
  dt16[g] = (r < DT_RANK) ? (_Float16)dbc[(size_t)tok * 48 + r] : (_Float16)0.f;
}

__global__ void rmsnorm_kernel(const float* __restrict__ h, const float* __restrict__ w,
                               _Float16* __restrict__ out) {
  __shared__ float red[D_MODEL];
  const int tok = blockIdx.x, tid = threadIdx.x;
  float v = h[(size_t)tok * D_MODEL + tid];
  red[tid] = v * v;
  __syncthreads();
  for (int s = D_MODEL / 2; s > 0; s >>= 1) {
    if (tid < s) red[tid] += red[tid + s];
    __syncthreads();
  }
  float scale = rsqrtf(red[0] * (1.f / D_MODEL) + 1e-5f);
  out[(size_t)tok * D_MODEL + tid] = (_Float16)(v * scale * w[tid]);
}

// Causal depthwise conv (K=4) + bias + SiLU.  xb lives in first ED cols of xz.
__global__ void conv_kernel(const float* __restrict__ xz, const float* __restrict__ cw,
                            const float* __restrict__ cb, float* __restrict__ xb_f32,
                            _Float16* __restrict__ xb_f16) {
  int g = blockIdx.x * blockDim.x + threadIdx.x;
  if (g >= NTOK * ED) return;
  const int e   = g & (ED - 1);
  const int tok = g >> 9;               // /ED
  const int t   = tok & (SEQ_L - 1);
  const int b   = tok >> 11;            // /SEQ_L
  float acc = cb[e];
  #pragma unroll
  for (int j = 0; j < D_CONV; ++j) {
    int tt = t - (D_CONV - 1) + j;
    if (tt >= 0)
      acc += xz[(size_t)(b * SEQ_L + tt) * (2 * ED) + e] * cw[e * D_CONV + j];
  }
  float s = acc * sigmoid_f(acc);
  xb_f32[g] = s;
  xb_f16[g] = (_Float16)s;
}

// Selective scan: one thread per channel e, 16-state recurrence in VGPRs,
// fused with D*x skip and SiLU(z) gate; emits f16 for the out_proj WMMA GEMM.
// global_prefetch issued 16 tokens ahead to hide L2 latency on the serial path.
__global__ void scan_kernel(const float* __restrict__ delta, const float* __restrict__ xb,
                            const float* __restrict__ xz,    const float* __restrict__ dbc,
                            const float* __restrict__ A_log, const float* __restrict__ Dp,
                            _Float16* __restrict__ y16) {
  const int b = blockIdx.x >> 2;
  const int e = ((blockIdx.x & 3) * 128) + threadIdx.x;
  float Ar[N_STATE], h[N_STATE];
  #pragma unroll
  for (int n = 0; n < N_STATE; ++n) {
    Ar[n] = -expf(A_log[(size_t)e * N_STATE + n]);
    h[n]  = 0.f;
  }
  const float D = Dp[e];
  for (int t = 0; t < SEQ_L; ++t) {
    const size_t tok = (size_t)b * SEQ_L + t;
    if (t + 16 < SEQ_L) {
      const size_t ptok = tok + 16;
      if ((threadIdx.x & 31) == 0)
        __builtin_prefetch(dbc + ptok * 48 + 16, 0, 0);   // B/C broadcast line
      __builtin_prefetch(delta + ptok * ED + e, 0, 0);    // per-lane streams
      __builtin_prefetch(xb + ptok * ED + e, 0, 0);
    }
    const float d = delta[tok * ED + e];
    const float x = xb[tok * ED + e];
    const float z = xz[tok * 2 * ED + ED + e];
    const float4* Bp = (const float4*)(dbc + tok * 48 + 16);
    const float4* Cp = (const float4*)(dbc + tok * 48 + 32);
    float Bm[N_STATE], Cm[N_STATE];
    #pragma unroll
    for (int q = 0; q < 4; ++q) {
      float4 bv = Bp[q], cv = Cp[q];
      Bm[4*q+0] = bv.x; Bm[4*q+1] = bv.y; Bm[4*q+2] = bv.z; Bm[4*q+3] = bv.w;
      Cm[4*q+0] = cv.x; Cm[4*q+1] = cv.y; Cm[4*q+2] = cv.z; Cm[4*q+3] = cv.w;
    }
    const float dx = d * x;
    float y = 0.f;
    #pragma unroll
    for (int n = 0; n < N_STATE; ++n) {
      float dA = expf(d * Ar[n]);
      h[n] = dA * h[n] + dx * Bm[n];
      y += h[n] * Cm[n];
    }
    float out = (y + D * x) * (z * sigmoid_f(z));
    y16[tok * ED + e] = (_Float16)out;
  }
}

__global__ void decode_kernel(const float* __restrict__ h, const float* __restrict__ dw,
                              const float* __restrict__ db, float* __restrict__ out) {
  __shared__ float red[D_MODEL];
  const int b = blockIdx.x, tid = threadIdx.x;
  const size_t tok = (size_t)b * SEQ_L + (SEQ_L - 1);
  red[tid] = h[tok * D_MODEL + tid] * dw[tid];
  __syncthreads();
  for (int s = D_MODEL / 2; s > 0; s >>= 1) {
    if (tid < s) red[tid] += red[tid + s];
    __syncthreads();
  }
  if (tid == 0) out[b] = sigmoid_f(red[0] + db[0]);
}

// ---------------------------------------------------------------------------
static inline int gemm_blocks(int M, int N) {
  long long waves = (long long)(M / (16 * MT)) * (N / 16);
  return (int)((waves * 32 + 255) / 256);
}

extern "C" void kernel_launch(void* const* d_in, const int* in_sizes, int n_in,
                              void* d_out, int out_size, void* d_ws, size_t ws_size,
                              hipStream_t stream) {
  (void)in_sizes; (void)n_in; (void)out_size; (void)ws_size;
  const float* x        = (const float*)d_in[0];
  const float* enc_w    = (const float*)d_in[1];
  const float* enc_b    = (const float*)d_in[2];
  const float* norm_w   = (const float*)d_in[3];
  const float* in_projw = (const float*)d_in[4];
  const float* conv_w   = (const float*)d_in[5];
  const float* conv_b   = (const float*)d_in[6];
  const float* x_projw  = (const float*)d_in[7];
  const float* dt_projw = (const float*)d_in[8];
  const float* dt_projb = (const float*)d_in[9];
  const float* A_log    = (const float*)d_in[10];
  const float* D_param  = (const float*)d_in[11];
  const float* out_projw= (const float*)d_in[12];
  const float* dec_w    = (const float*)d_in[13];
  const float* dec_b    = (const float*)d_in[14];
  float* out = (float*)d_out;

  char* ws = (char*)d_ws;
  size_t off = 0;
  auto alloc = [&](size_t bytes) -> void* {
    void* p = ws + off;
    off = (off + bytes + 255) & ~(size_t)255;
    return p;
  };
  float*     hbuf  = (float*)    alloc((size_t)NTOK * D_MODEL * 4);
  _Float16*  x16   = (_Float16*) alloc((size_t)NTOK * IN_DIM * 2);
  _Float16*  xn16  = (_Float16*) alloc((size_t)NTOK * D_MODEL * 2);
  float*     xzb   = (float*)    alloc((size_t)NTOK * 2 * ED * 4);
  float*     xbf   = (float*)    alloc((size_t)NTOK * ED * 4);
  _Float16*  xb16  = (_Float16*) alloc((size_t)NTOK * ED * 2);   // reused as y16
  float*     dbc   = (float*)    alloc((size_t)NTOK * 48 * 4);
  _Float16*  dt16  = (_Float16*) alloc((size_t)NTOK * 32 * 2);
  float*     delta = (float*)    alloc((size_t)NTOK * ED * 4);
  _Float16*  enc16 = (_Float16*) alloc((size_t)D_MODEL * IN_DIM * 2);
  _Float16*  inp16 = (_Float16*) alloc((size_t)N_LAYERS * 2 * ED * D_MODEL * 2);
  _Float16*  xp16  = (_Float16*) alloc((size_t)N_LAYERS * 48 * ED * 2);
  _Float16*  dtw16 = (_Float16*) alloc((size_t)N_LAYERS * ED * 32 * 2);
  _Float16*  op16  = (_Float16*) alloc((size_t)N_LAYERS * D_MODEL * ED * 2);

  auto cvt = [&](const float* s, _Float16* d, int n) {
    f32_to_f16_kernel<<<(n + 255) / 256, 256, 0, stream>>>(s, d, n);
  };
  // one-time f16 weight/input conversion
  cvt(x, x16, NTOK * IN_DIM);
  cvt(enc_w, enc16, D_MODEL * IN_DIM);
  cvt(in_projw, inp16, N_LAYERS * 2 * ED * D_MODEL);
  cvt(x_projw, xp16, N_LAYERS * 48 * ED);
  cvt(out_projw, op16, N_LAYERS * D_MODEL * ED);
  pad_f16_kernel<<<(N_LAYERS * ED * 32 + 255) / 256, 256, 0, stream>>>(
      dt_projw, dtw16, N_LAYERS * ED, DT_RANK, 32);

  // encoder: h = x @ enc_w^T + enc_b
  wmma_gemm_kernel<1><<<gemm_blocks(NTOK, D_MODEL), 256, 0, stream>>>(
      x16, enc16, hbuf, enc_b, NTOK, D_MODEL, IN_DIM);

  for (int l = 0; l < N_LAYERS; ++l) {
    rmsnorm_kernel<<<NTOK, D_MODEL, 0, stream>>>(hbuf, norm_w + l * D_MODEL, xn16);
    // xz = xn @ in_proj^T
    wmma_gemm_kernel<0><<<gemm_blocks(NTOK, 2 * ED), 256, 0, stream>>>(
        xn16, inp16 + (size_t)l * 2 * ED * D_MODEL, xzb, nullptr, NTOK, 2 * ED, D_MODEL);
    // causal depthwise conv + SiLU
    conv_kernel<<<(NTOK * ED + 255) / 256, 256, 0, stream>>>(
        xzb, conv_w + (size_t)l * ED * D_CONV, conv_b + (size_t)l * ED, xbf, xb16);
    // dbc = xb @ x_proj^T   (dt | B | C)
    wmma_gemm_kernel<0><<<gemm_blocks(NTOK, 48), 256, 0, stream>>>(
        xb16, xp16 + (size_t)l * 48 * ED, dbc, nullptr, NTOK, 48, ED);
    dtprep_kernel<<<(NTOK * 32 + 255) / 256, 256, 0, stream>>>(dbc, dt16);
    // delta = softplus(dt @ dt_proj^T + dt_proj_b)  (K padded 16->32)
    wmma_gemm_kernel<2><<<gemm_blocks(NTOK, ED), 256, 0, stream>>>(
        dt16, dtw16 + (size_t)l * ED * 32, delta, dt_projb + (size_t)l * ED, NTOK, ED, 32);
    // selective scan fused with D*x skip and SiLU(z) gate -> y16 (reuses xb16)
    scan_kernel<<<B_SZ * 4, 128, 0, stream>>>(
        delta, xbf, xzb, dbc, A_log + (size_t)l * ED * N_STATE, D_param + (size_t)l * ED, xb16);
    // h += y @ out_proj^T  (residual fused into WMMA epilogue)
    wmma_gemm_kernel<3><<<gemm_blocks(NTOK, D_MODEL), 256, 0, stream>>>(
        xb16, op16 + (size_t)l * D_MODEL * ED, hbuf, nullptr, NTOK, D_MODEL, ED);
  }

  decode_kernel<<<B_SZ, D_MODEL, 0, stream>>>(hbuf, dec_w, dec_b, out);
}
